// PESP_64484638982894
// MI455X (gfx1250) — matrix operand order repaired and checked by reading, not compiled, not run on hardware
//
#include <hip/hip_runtime.h>
#include <hip/hip_bf16.h>

typedef __attribute__((ext_vector_type(16))) _Float16 v16h;
typedef __attribute__((ext_vector_type(8)))  float    v8f;

#define NX   48
#define VOX  110592      // 48^3
#define NF   64
#define NF3  262144      // 64^3

static __device__ __forceinline__ int clampNX(int v) {
    v = v < 0 ? 0 : v;               // lowers to v_med3_i32
    return v > (NX - 1) ? (NX - 1) : v;
}

// ---------------------------------------------------------------------------
// Weight pre-pack: emit A-fragments as 16x32 f16 WMMA lanes in contiguous k order (k = 16*hi + e),
// zero-padded, so conv kernels load A with unconditional 32B vector loads.
//   mode 0: 1x1 conv, K = input channels        (nchunk = Cin/32)
//   mode 1: 3^3 conv Cin<=16, K = 2 taps x 16ch (nchunk = 14)
//   mode 2: 3^3 conv Cin==1,  K = 27 taps pad 32 (nchunk = 1)
// Packed index: (((mt*nchunk + q)*32 + lane)*16 + e)
// ---------------------------------------------------------------------------
__global__ __launch_bounds__(256)
void pesp_pack_a(const float* __restrict__ W, _Float16* __restrict__ out,
                 int Cin, int Cout, int nchunk, int mode, int total)
{
    int i = blockIdx.x * blockDim.x + threadIdx.x;
    if (i >= total) return;
    int e    = i & 15;
    int lane = (i >> 4) & 31;
    int q    = (i >> 9) % nchunk;
    int mt   = (i >> 9) / nchunk;
    int m    = mt * 16 + (lane & 15);
    int hi   = lane >> 4;
    int k    = e + hi * 16;                        // K within chunk, A-layout
    float v = 0.f;
    if (mode == 0) {
        int ci = q * 32 + k;
        if (m < Cout && ci < Cin) v = W[m * Cin + ci];
    } else if (mode == 1) {
        int tap = q * 2 + (k >> 4), c = k & 15;
        if (m < Cout && tap < 27 && c < Cin) v = W[(m * Cin + c) * 27 + tap];
    } else {
        int tap = k;
        if (m < Cout && tap < 27) v = W[m * 27 + tap];
    }
    out[i] = (_Float16)v;
}

// ---------------------------------------------------------------------------
// 1x1 conv as GEMM: D[Co,vox] = sum_Ci W[Co,Ci] * X[Ci,vox].  Cin fixed = 64.
// One wave -> 16 out-channels x 2x16 voxels; A fragments reused across both
// N-tiles (4 WMMAs / wave).  Output buffers are channel-padded to multiples
// of 16 so all stores are unconditional.
// ---------------------------------------------------------------------------
template <bool RELU, bool F32OUT>
__global__ __launch_bounds__(128)
void pesp_conv1x1_wmma(const _Float16* __restrict__ Apack, const _Float16* __restrict__ X,
                       void* __restrict__ Yv)
{
    const int lane  = threadIdx.x & 31;
    const int mt    = threadIdx.x >> 5;
    const int hi    = lane >> 4;
    const int n     = lane & 15;
    const int vbase = blockIdx.x * 32 + n;        // two 16-voxel tiles

    __builtin_prefetch(&Apack[(size_t)mt * 2 * 512], 0, 0);

    v8f acc0 = {}, acc1 = {};
#pragma unroll
    for (int q = 0; q < 2; ++q) {
        v16h a = *(const v16h*)(Apack + (((size_t)mt * 2 + q) * 32 + lane) * 16);
        v16h b0, b1;
#pragma unroll
        for (int e = 0; e < 16; ++e) {
            size_t row = (size_t)(q * 32 + e + hi * 16) * VOX;
            b0[e] = X[row + vbase];
            b1[e] = X[row + vbase + 16];
        }
        acc0 = __builtin_amdgcn_wmma_f32_16x16x32_f16(false, a, false, b0,
                                                      (short)0, acc0, false, false);
        acc1 = __builtin_amdgcn_wmma_f32_16x16x32_f16(false, a, false, b1,
                                                      (short)0, acc1, false, false);
    }
#pragma unroll
    for (int r = 0; r < 8; ++r) {
        int mo = mt * 16 + r + hi * 8;
        float v0 = acc0[r], v1 = acc1[r];
        if (RELU) { v0 = v0 > 0.f ? v0 : 0.f; v1 = v1 > 0.f ? v1 : 0.f; }
        if (F32OUT) {
            float* Y = (float*)Yv;
            Y[(size_t)mo * VOX + vbase]      = v0;
            Y[(size_t)mo * VOX + vbase + 16] = v1;
        } else {
            _Float16* Y = (_Float16*)Yv;
            Y[(size_t)mo * VOX + vbase]      = (_Float16)v0;
            Y[(size_t)mo * VOX + vbase + 16] = (_Float16)v1;
        }
    }
}

// ---------------------------------------------------------------------------
// 3x3x3 dilated conv, implicit GEMM, A from pack (zero-padded).
// Branch-free B gather: per-coordinate med3 clamp -> unconditional address +
// load, result zeroed by cndmask where out-of-bounds.  tap/c padding positions
// map to weight==0, so their B values are don't-cares (clamped to valid data).
// Dilations up to 16 make LDS halo tiling a bad fit; the 3.5MB f16 input map
// is L2-resident on MI455X (192MB), so global gathers are L2-served.
// A fragments reused across 2 N-tiles.
// ---------------------------------------------------------------------------
template <int CIN, bool RELU>
__global__ __launch_bounds__(128)
void pesp_conv3_wmma(const _Float16* __restrict__ Apack, const _Float16* __restrict__ X,
                     _Float16* __restrict__ Y, int dil)
{
    const int lane  = threadIdx.x & 31;
    const int mt    = threadIdx.x >> 5;
    const int hi    = lane >> 4;
    const int n     = lane & 15;
    const int vbase = blockIdx.x * 32 + n;

    constexpr int NCHUNK = (CIN == 1) ? 1 : 14;
    __builtin_prefetch(&Apack[(size_t)mt * NCHUNK * 512], 0, 0);

    v8f acc0 = {}, acc1 = {};
#pragma unroll
    for (int q = 0; q < NCHUNK; ++q) {
        v16h a = *(const v16h*)(Apack + (((size_t)mt * NCHUNK + q) * 32 + lane) * 16);
#pragma unroll
        for (int t = 0; t < 2; ++t) {
            const int vox = vbase + t * 16;
            const int iz  = vox % NX;
            const int iy  = (vox / NX) % NX;
            const int ix  = vox / (NX * NX);
            v16h b;
#pragma unroll
            for (int e = 0; e < 16; ++e) {
                int k   = e + hi * 16;                       // B: 32x16 layout
                int tap = (CIN == 1) ? k : q * 2 + (k >> 4);
                int c   = (CIN == 1) ? 0 : (k & 15);
                if (tap > 26) tap = 13;                      // weight==0 there
                if (c >= CIN) c = 0;                         // weight==0 there
                int jx = ix + (tap / 9 - 1) * dil;
                int jy = iy + ((tap / 3) % 3 - 1) * dil;
                int jz = iz + (tap % 3 - 1) * dil;
                bool inb = (unsigned)jx < NX && (unsigned)jy < NX && (unsigned)jz < NX;
                int sp = (clampNX(jx) * NX + clampNX(jy)) * NX + clampNX(jz);
                _Float16 xv = X[(size_t)c * VOX + sp];       // unconditional load
                b[e] = inb ? xv : (_Float16)0.f;             // cndmask, no exec churn
            }
            if (t == 0)
                acc0 = __builtin_amdgcn_wmma_f32_16x16x32_f16(false, a, false, b,
                                                              (short)0, acc0, false, false);
            else
                acc1 = __builtin_amdgcn_wmma_f32_16x16x32_f16(false, a, false, b,
                                                              (short)0, acc1, false, false);
        }
    }
#pragma unroll
    for (int r = 0; r < 8; ++r) {
        int mo = mt * 16 + r + hi * 8;
        float v0 = acc0[r], v1 = acc1[r];
        if (RELU) { v0 = v0 > 0.f ? v0 : 0.f; v1 = v1 > 0.f ? v1 : 0.f; }
        Y[(size_t)mo * VOX + vbase]      = (_Float16)v0;
        Y[(size_t)mo * VOX + vbase + 16] = (_Float16)v1;
    }
}

// ---------------------------------------------------------------------------
// ESP merge: concat[d1, d2, d2+d4, d2+d4+d8, d2+d4+d8+d16] + x, BN(eval), PReLU
// ---------------------------------------------------------------------------
__global__ __launch_bounds__(256)
void pesp_esp_merge(const _Float16* __restrict__ d1, const _Float16* __restrict__ d2,
                    const _Float16* __restrict__ d4, const _Float16* __restrict__ d8,
                    const _Float16* __restrict__ d16, const _Float16* __restrict__ xin,
                    const float* __restrict__ bn_beta, const float* __restrict__ bn_gamma,
                    const float* __restrict__ bn_mean, const float* __restrict__ bn_var,
                    const float* __restrict__ prelu, _Float16* __restrict__ out)
{
    size_t i = (size_t)blockIdx.x * blockDim.x + threadIdx.x;
    if (i >= (size_t)64 * VOX) return;
    int c = (int)(i / VOX);
    int v = (int)(i % VOX);
    float val;
    if (c < 16) {
        val = (float)d1[(size_t)c * VOX + v];
    } else {
        int g  = (c - 16) / 12;
        int cc = (c - 16) % 12;
        size_t off = (size_t)cc * VOX + v;
        val = (float)d2[off];
        if (g >= 1) val += (float)d4[off];
        if (g >= 2) val += (float)d8[off];
        if (g >= 3) val += (float)d16[off];
    }
    val += (float)xin[i];
    val = (val - bn_mean[c]) * rsqrtf(bn_var[c] + 1e-3f) * bn_gamma[c] + bn_beta[c];
    float a = prelu[c];
    out[i] = (_Float16)(val >= 0.f ? val : a * val);
}

// ---------------------------------------------------------------------------
// FFT machinery: in-place 64-point DFT along one axis (one line per block).
// ---------------------------------------------------------------------------
__global__ __launch_bounds__(64)
void pesp_dft64(float* __restrict__ re, float* __restrict__ im,
                int sa, int s1, int s2, int sign, float scale)
{
    __shared__ float lr[64], li[64];
    int t   = threadIdx.x;
    int o1  = blockIdx.x & 63;
    int o2  = blockIdx.x >> 6;
    int idx = o1 * s1 + o2 * s2 + t * sa;
    lr[t] = re[idx];
    li[t] = im[idx];
    __syncthreads();
    float ar = 0.f, ai = 0.f;
    const float w0 = (float)sign * 6.283185307179586f / 64.f;
    for (int k = 0; k < 64; ++k) {
        float s, c;
        __sincosf(w0 * (float)((t * k) & 63), &s, &c);
        ar += lr[k] * c - li[k] * s;
        ai += lr[k] * s + li[k] * c;
    }
    re[idx] = ar * scale;
    im[idx] = ai * scale;
}

__global__ __launch_bounds__(256)
void pesp_fft_pad(const float* __restrict__ x, float* __restrict__ re, float* __restrict__ im)
{
    int i = blockIdx.x * blockDim.x + threadIdx.x;
    if (i >= NF3) return;
    int iz = i & 63, iy = (i >> 6) & 63, ix = i >> 12;
    float v = 0.f;
    if (ix < NX && iy < NX && iz < NX) v = x[(ix * NX + iy) * NX + iz];
    re[i] = v;
    im[i] = 0.f;
}

__global__ __launch_bounds__(256)
void pesp_spec_mul(float* __restrict__ re, float* __restrict__ im,
                   const float* __restrict__ dk, int square)
{
    int i = blockIdx.x * blockDim.x + threadIdx.x;
    if (i >= NF3) return;
    float d = dk[i];
    if (square) d *= d;
    re[i] *= d;
    im[i] *= d;
}

__global__ __launch_bounds__(256)
void pesp_fft_crop(const float* __restrict__ re, const float* __restrict__ alpha,
                   float* __restrict__ out)
{
    int v = blockIdx.x * blockDim.x + threadIdx.x;
    if (v >= VOX) return;
    int iz = v % NX, iy = (v / NX) % NX, ix = v / (NX * NX);
    out[v] = alpha[0] * re[(ix * NF + iy) * NF + iz];
}

// x_in = (x_est + pn - mean)/std * mask   (pn = den - deconv(den))
__global__ __launch_bounds__(256)
void pesp_prep_xin(const float* __restrict__ x_est, const float* __restrict__ den,
                   const float* __restrict__ dcv, const float* __restrict__ mask,
                   const float* __restrict__ mean, const float* __restrict__ stdv,
                   _Float16* __restrict__ xin, int have_pn)
{
    int v = blockIdx.x * blockDim.x + threadIdx.x;
    if (v >= VOX) return;
    float pn = have_pn ? (den[v] - dcv[v]) : 0.f;
    xin[v] = (_Float16)((x_est[v] + pn - mean[0]) / stdv[0] * mask[v]);
}

// den = (x_pred*std + mean)*mask ; out[iter] = x_pred
__global__ __launch_bounds__(256)
void pesp_den_out(const float* __restrict__ x_pred, const float* __restrict__ mask,
                  const float* __restrict__ mean, const float* __restrict__ stdv,
                  float* __restrict__ den, float* __restrict__ out_slice)
{
    int v = blockIdx.x * blockDim.x + threadIdx.x;
    if (v >= VOX) return;
    float xp = x_pred[v];
    den[v]       = (xp * stdv[0] + mean[0]) * mask[v];
    out_slice[v] = xp;
}

// ---------------------------------------------------------------------------
extern "C" void kernel_launch(void* const* d_in, const int* in_sizes, int n_in,
                              void* d_out, int out_size, void* d_ws, size_t ws_size,
                              hipStream_t stream)
{
    const float* y     = (const float*)d_in[0];
    const float* dk    = (const float*)d_in[1];
    const float* mask  = (const float*)d_in[2];
    const float* alpha = (const float*)d_in[3];
    const float* gmean = (const float*)d_in[4];
    const float* gstd  = (const float*)d_in[5];

    // jax pytree flatten sorts dict keys: params -> blocks(5 x 11 sorted), conv0, fc1, fc2, fc3
    // per-block sorted: bn_beta, bn_gamma, bn_mean, bn_var, c1, d1, d16, d2, d4, d8, prelu
    struct Blk { const float *bn_beta,*bn_gamma,*bn_mean,*bn_var,*c1,*d1,*d16,*d2,*d4,*d8,*prelu; };
    Blk blk[5];
    int p = 6;
    for (int b = 0; b < 5; ++b) {
        blk[b].bn_beta  = (const float*)d_in[p++];
        blk[b].bn_gamma = (const float*)d_in[p++];
        blk[b].bn_mean  = (const float*)d_in[p++];
        blk[b].bn_var   = (const float*)d_in[p++];
        blk[b].c1       = (const float*)d_in[p++];
        blk[b].d1       = (const float*)d_in[p++];
        blk[b].d16      = (const float*)d_in[p++];
        blk[b].d2       = (const float*)d_in[p++];
        blk[b].d4       = (const float*)d_in[p++];
        blk[b].d8       = (const float*)d_in[p++];
        blk[b].prelu    = (const float*)d_in[p++];
    }
    const float* conv0 = (const float*)d_in[p++];
    const float* fc1   = (const float*)d_in[p++];
    const float* fc2   = (const float*)d_in[p++];
    const float* fc3   = (const float*)d_in[p++];

    // ---- workspace carve (~62 MB) ----
    char* w = (char*)d_ws;
    auto carve = [&](size_t bytes) -> char* {
        char* r = w; w += (bytes + 255) & ~(size_t)255; return r;
    };
    float*     re     = (float*)carve(NF3 * 4);
    float*     im     = (float*)carve(NF3 * 4);
    float*     x_est  = (float*)carve(VOX * 4);
    float*     dcv    = (float*)carve(VOX * 4);
    float*     den    = (float*)carve(VOX * 4);
    float*     x_pred = (float*)carve((size_t)16 * VOX * 4);   // padded: fc3 stores 16 rows
    _Float16*  xin    = (_Float16*)carve(VOX * 2);
    _Float16*  h0     = (_Float16*)carve((size_t)64 * VOX * 2);
    _Float16*  h1     = (_Float16*)carve((size_t)64 * VOX * 2);
    _Float16*  o1     = (_Float16*)carve((size_t)16 * VOX * 2);
    _Float16*  db[5];
    for (int i = 0; i < 5; ++i) db[i] = (_Float16*)carve((size_t)16 * VOX * 2);

    // packed weights
    auto pack = [&](const float* W, int Cin, int Cout, int nchunk, int mode) -> _Float16* {
        int mtiles = (Cout + 15) / 16;
        int total  = mtiles * nchunk * 512;
        _Float16* dst = (_Float16*)carve((size_t)total * 2);
        pesp_pack_a<<<(total + 255) / 256, 256, 0, stream>>>(W, dst, Cin, Cout, nchunk, mode, total);
        return dst;
    };
    _Float16* P_conv0 = pack(conv0, 1, 64, 1, 2);
    _Float16* P_fc1   = pack(fc1, 64, 64, 2, 0);
    _Float16* P_fc2   = pack(fc2, 64, 64, 2, 0);
    _Float16* P_fc3   = pack(fc3, 64, 1, 2, 0);
    _Float16 *P_c1[5], *P_d1[5], *P_d2[5], *P_d4[5], *P_d8[5], *P_d16[5];
    for (int b = 0; b < 5; ++b) {
        P_c1[b]  = pack(blk[b].c1, 64, 12, 2, 0);
        P_d1[b]  = pack(blk[b].d1, 12, 16, 14, 1);
        P_d2[b]  = pack(blk[b].d2, 12, 12, 14, 1);
        P_d4[b]  = pack(blk[b].d4, 12, 12, 14, 1);
        P_d8[b]  = pack(blk[b].d8, 12, 12, 14, 1);
        P_d16[b] = pack(blk[b].d16, 12, 12, 14, 1);
    }

    const int VT2 = VOX / 32;                      // 3456 dual-tile blocks
    const int E   = (VOX + 255) / 256;
    const int EF  = (NF3 + 255) / 256;
    const int EM  = (64 * VOX + 255) / 256;

    auto dft_all = [&](int sign, float scale) {
        pesp_dft64<<<4096, 64, 0, stream>>>(re, im,    1, 64, 4096, sign, scale);
        pesp_dft64<<<4096, 64, 0, stream>>>(re, im,   64,  1, 4096, sign, scale);
        pesp_dft64<<<4096, 64, 0, stream>>>(re, im, 4096,  1,   64, sign, scale);
    };
    auto deconv_run = [&](const float* src, int square, float* dst) {
        pesp_fft_pad<<<EF, 256, 0, stream>>>(src, re, im);
        dft_all(-1, 1.f);
        pesp_spec_mul<<<EF, 256, 0, stream>>>(re, im, dk, square);
        dft_all(+1, 1.f / 64.f);
        pesp_fft_crop<<<E, 256, 0, stream>>>(re, alpha, dst);
    };

    deconv_run(y, 0, x_est);

    float* out = (float*)d_out;
    for (int it = 0; it < 3; ++it) {
        if (it == 0) {
            pesp_prep_xin<<<E, 256, 0, stream>>>(x_est, den, dcv, mask, gmean, gstd, xin, 0);
        } else {
            deconv_run(den, 1, dcv);
            pesp_prep_xin<<<E, 256, 0, stream>>>(x_est, den, dcv, mask, gmean, gstd, xin, 1);
        }

        // --- gen ---
        pesp_conv3_wmma<1, true><<<VT2, 128, 0, stream>>>(P_conv0, xin, h0, 1);
        _Float16 *cur = h0, *alt = h1;
        for (int b = 0; b < 5; ++b) {
            pesp_conv1x1_wmma<false, false><<<VT2, 32, 0, stream>>>(P_c1[b], cur, o1);
            pesp_conv3_wmma<12, false><<<VT2, 32, 0, stream>>>(P_d1[b],  o1, db[0],  1);
            pesp_conv3_wmma<12, false><<<VT2, 32, 0, stream>>>(P_d2[b],  o1, db[1],  2);
            pesp_conv3_wmma<12, false><<<VT2, 32, 0, stream>>>(P_d4[b],  o1, db[2],  4);
            pesp_conv3_wmma<12, false><<<VT2, 32, 0, stream>>>(P_d8[b],  o1, db[3],  8);
            pesp_conv3_wmma<12, false><<<VT2, 32, 0, stream>>>(P_d16[b], o1, db[4], 16);
            pesp_esp_merge<<<EM, 256, 0, stream>>>(db[0], db[1], db[2], db[3], db[4], cur,
                                                   blk[b].bn_beta, blk[b].bn_gamma,
                                                   blk[b].bn_mean, blk[b].bn_var,
                                                   blk[b].prelu, alt);
            _Float16* t = cur; cur = alt; alt = t;
        }
        pesp_conv1x1_wmma<true, false><<<VT2, 128, 0, stream>>>(P_fc1, cur, alt);
        { _Float16* t = cur; cur = alt; alt = t; }
        pesp_conv1x1_wmma<true, false><<<VT2, 128, 0, stream>>>(P_fc2, cur, alt);
        { _Float16* t = cur; cur = alt; alt = t; }
        pesp_conv1x1_wmma<false, true><<<VT2, 32, 0, stream>>>(P_fc3, cur, x_pred);

        pesp_den_out<<<E, 256, 0, stream>>>(x_pred, mask, gmean, gstd, den,
                                            out + (size_t)it * VOX);
    }
}
